// GroupedQueryAttention_75445395522220
// MI455X (gfx1250) — matrix operand order repaired and checked by reading, not compiled, not run on hardware
//
#include <hip/hip_runtime.h>
#include <hip/hip_bf16.h>

typedef __bf16 bf16_t;
typedef __attribute__((ext_vector_type(16))) __bf16 v16bf;
typedef __attribute__((ext_vector_type(8)))  __bf16 v8bf;
typedef __attribute__((ext_vector_type(8)))  float  v8f;

#define T_SEQ 2048

// Load a 16-element bf16 fragment from two 8-element (16-byte) chunks.
__device__ __forceinline__ v16bf load_frag16(const bf16_t* p0, const bf16_t* p1) {
    v8bf lo = *(const v8bf*)p0;
    v8bf hi = *(const v8bf*)p1;
    v16bf r;
#pragma unroll
    for (int i = 0; i < 8; ++i) { r[i] = lo[i]; r[i + 8] = hi[i]; }
    return r;
}

__device__ __forceinline__ v8f wmma_bf16(v16bf a, v16bf b, v8f c) {
    // (neg_a, A, neg_b, B, c_mod, C, reuse_a, reuse_b)
    return __builtin_amdgcn_wmma_f32_16x16x32_bf16(false, a, false, b, (short)0, c,
                                                   false, false);
}

// ---------------------------------------------------------------------------
// fp32 -> bf16 elementwise conversion
// ---------------------------------------------------------------------------
__global__ void cvt_copy(const float* __restrict__ src, bf16_t* __restrict__ dst, int n) {
    int i = blockIdx.x * blockDim.x + threadIdx.x;
    if (i < n) dst[i] = (bf16_t)src[i];
}

// W: [K][N] fp32 row-major -> Wt: [N][K] bf16, 32x32 LDS-tiled (coalesced writes).
__global__ __launch_bounds__(256)
void cvt_transpose_tiled(const float* __restrict__ W, bf16_t* __restrict__ Wt,
                         int K, int N) {
    __shared__ bf16_t tile[32][33];
    const int tx = threadIdx.x, ty = threadIdx.y;
    const int kt = blockIdx.y * 32, nt = blockIdx.x * 32;
#pragma unroll
    for (int j = 0; j < 4; ++j)
        tile[ty + j * 8][tx] = (bf16_t)W[(size_t)(kt + ty + j * 8) * N + nt + tx];
    __syncthreads();
#pragma unroll
    for (int j = 0; j < 4; ++j)
        Wt[(size_t)(nt + ty + j * 8) * K + kt + tx] = tile[tx][ty + j * 8];
}

// Concatenated QKV bias: [0,2048)=bq, [2048,2560)=bk, [2560,3072)=bv
__global__ void concat_bias(const float* __restrict__ bq, const float* __restrict__ bk,
                            const float* __restrict__ bv, float* __restrict__ dst) {
    int i = blockIdx.x * blockDim.x + threadIdx.x;
    if (i < 2048)      dst[i] = bq[i];
    else if (i < 2560) dst[i] = bk[i - 2048];
    else if (i < 3072) dst[i] = bv[i - 2560];
}

// ---------------------------------------------------------------------------
// bf16 WMMA GEMM:  C[m,n] = sum_k A[m][k] * Bt[n][k] + bias[n]
// Block = 256 threads = 8 waves (4(M) x 2(N)); wave tile 64x64 = 4x4 WMMA accs.
// Block tile = 256(M) x 128(N).
// MODE 0: fp32 row-major out. MODE 1: fused QKV split epilogue (N=3072).
// ---------------------------------------------------------------------------
template <int MODE>
__global__ __launch_bounds__(256)
void gemm_bf16(const bf16_t* __restrict__ A, const bf16_t* __restrict__ Bt,
               const float* __restrict__ bias, int M, int N, int K,
               float* __restrict__ outf, bf16_t* __restrict__ outQ,
               bf16_t* __restrict__ outK, bf16_t* __restrict__ outV) {
    const int lane = threadIdx.x & 31;
    const int wave = threadIdx.x >> 5;
    const int wm = wave >> 1, wn = wave & 1;
    const int m0 = blockIdx.y * 256 + wm * 64;
    const int n0 = blockIdx.x * 128 + wn * 64;
    const int l15 = lane & 15;
    const int kbA = (lane < 16) ? 0 : 8;
    const int kbB = (lane < 16) ? 0 : 8;

    v8f acc[4][4];
#pragma unroll
    for (int i = 0; i < 4; ++i)
#pragma unroll
        for (int j = 0; j < 4; ++j) acc[i][j] = {};

    const bf16_t* arow = A + (size_t)(m0 + l15) * K;
    const bf16_t* bcol = Bt + (size_t)(n0 + l15) * K;

    for (int kw = 0; kw < K; kw += 32) {
        if (kw + 256 < K) {  // gfx1250 global_prefetch_b8, 8 k-steps ahead
            __builtin_prefetch(arow + kw + 256, 0, 0);
            __builtin_prefetch(bcol + kw + 256, 0, 0);
        }
        v16bf af[4], bf[4];
#pragma unroll
        for (int i = 0; i < 4; ++i) {
            const bf16_t* p = arow + (size_t)i * 16 * K + kw + kbA;
            af[i] = load_frag16(p, p + 16);
        }
#pragma unroll
        for (int j = 0; j < 4; ++j) {
            const bf16_t* p = bcol + (size_t)j * 16 * K + kw + kbB;
            bf[j] = load_frag16(p, p + 16);
        }
#pragma unroll
        for (int i = 0; i < 4; ++i)
#pragma unroll
            for (int j = 0; j < 4; ++j)
                acc[i][j] = wmma_bf16(af[i], bf[j], acc[i][j]);
    }

    const int rbase = (lane >> 4) << 3;  // 0 or 8
#pragma unroll
    for (int i = 0; i < 4; ++i)
#pragma unroll
        for (int j = 0; j < 4; ++j) {
            int tn = n0 + j * 16 + l15;
            float bval = bias[tn];
#pragma unroll
            for (int r = 0; r < 8; ++r) {
                int m = m0 + i * 16 + rbase + r;
                float val = acc[i][j][r] + bval;
                if (MODE == 0) {
                    outf[(size_t)m * N + tn] = val;
                } else {
                    int b = m >> 11;            // T_SEQ = 2048
                    int t = m & (T_SEQ - 1);
                    if (tn < 2048) {            // Q: [B][32][T][64]
                        int h = tn >> 6, d = tn & 63;
                        outQ[((size_t)(b * 32 + h) * T_SEQ + t) * 64 + d] = (bf16_t)val;
                    } else if (tn < 2560) {     // K: [B][8][T][64]
                        int nk = tn - 2048;
                        int h = nk >> 6, d = nk & 63;
                        outK[((size_t)(b * 8 + h) * T_SEQ + t) * 64 + d] = (bf16_t)val;
                    } else {                    // V^T: [B][8][64][T]
                        int nv = tn - 2560;
                        int h = nv >> 6, d = nv & 63;
                        outV[((size_t)(b * 8 + h) * 64 + d) * T_SEQ + t] = (bf16_t)val;
                    }
                }
            }
        }
}

// ---------------------------------------------------------------------------
// Fused flash-style grouped-query attention, LDS-staged K/V.
//   Q : [B][32][T][64] bf16,  K : [B][8][T][64] bf16,  Vt : [B][8][64][T] bf16
//   AO: [B*T][2048] bf16
// Block = 256 threads = 8 waves; 128 Q rows per block (16 per wave). All waves
// share one kv head: K/V chunks (64 keys) are staged cooperatively into
// double-buffered LDS. All 8 B-fragments per phase are hoisted into distinct
// registers so the scheduler can clause the ds_loads and hide LDS latency
// under WMMA issue (staggered s_wait_dscnt instead of wait-0 per WMMA).
// ---------------------------------------------------------------------------
__global__ __launch_bounds__(256)
void attn_kernel(const bf16_t* __restrict__ Q, const bf16_t* __restrict__ Kt,
                 const bf16_t* __restrict__ Vt, bf16_t* __restrict__ AO) {
    __shared__ __align__(16) bf16_t kbuf[2][64 * 64];  // [s-local][d]
    __shared__ __align__(16) bf16_t vbuf[2][64 * 64];  // [d][s-local]
    __shared__ __align__(16) bf16_t pbuf[8][16 * 64];  // per-wave P tile

    const int tid  = threadIdx.x;
    const int lane = tid & 31;
    const int wave = tid >> 5;
    const int l15  = lane & 15;
    const int hi   = lane >> 4;
    const int kbA  = hi * 8;
    const int kbB  = hi * 8;
    const int rbase = hi * 8;

    const int t0  = blockIdx.x * 128 + wave * 16;
    const int bh  = blockIdx.y;        // b*32 + qh
    const int b   = bh >> 5;
    const int qh  = bh & 31;
    const int kvh = qh >> 2;           // 4 query heads per kv head

    const bf16_t* qrow = Q + ((size_t)bh * T_SEQ + t0 + l15) * 64;
    v16bf aq0 = load_frag16(qrow + kbA,      qrow + 16 + kbA);
    v16bf aq1 = load_frag16(qrow + 32 + kbA, qrow + 48 + kbA);

    const bf16_t* Kbase = Kt + (size_t)(b * 8 + kvh) * T_SEQ * 64;
    const bf16_t* Vbase = Vt + (size_t)(b * 8 + kvh) * 64 * T_SEQ;

    // Cooperative stage of one 64-key chunk (8KB K + 8KB V) into LDS.
    auto stage = [&](bf16_t* dk, bf16_t* dv, int s0) {
#pragma unroll
        for (int p = 0; p < 2; ++p) {
            int idx = p * 256 + tid;   // 0..511
            int row = idx >> 3;        // 0..63
            int seg = (idx & 7) * 8;   // 16B segments within 128B row
            const bf16_t* gk = Kbase + (size_t)(s0 + row) * 64 + seg;
            const bf16_t* gv = Vbase + (size_t)row * T_SEQ + s0 + seg;
            if (s0 + 128 < T_SEQ) {    // prefetch 2 chunks ahead
                __builtin_prefetch(gk + 128 * 64, 0, 0);
                __builtin_prefetch(gv + 128, 0, 0);
            }
            *(v8bf*)&dk[row * 64 + seg] = *(const v8bf*)gk;
            *(v8bf*)&dv[row * 64 + seg] = *(const v8bf*)gv;
        }
    };

    v8f o[4];
#pragma unroll
    for (int t = 0; t < 4; ++t) o[t] = {};
    float mrun[8], lrun[8];
#pragma unroll
    for (int r = 0; r < 8; ++r) { mrun[r] = -__builtin_inff(); lrun[r] = 0.0f; }

    bf16_t* myp = pbuf[wave];

    stage(kbuf[0], vbuf[0], 0);
    __syncthreads();

    for (int ci = 0; ci < T_SEQ / 64; ++ci) {
        const int s0 = ci * 64;
        if (ci + 1 < T_SEQ / 64)
            stage(kbuf[(ci + 1) & 1], vbuf[(ci + 1) & 1], s0 + 64);
        const bf16_t* kl = kbuf[ci & 1];
        const bf16_t* vl = vbuf[ci & 1];

        // --- scores: S[16 x 64] = Q_tile (16x64) . K_chunk^T (LDS) ---
        // Hoist all 8 B-fragments so ds_loads can be clause'd ahead of WMMAs.
        v16bf kf0[4], kf1[4];
#pragma unroll
        for (int c = 0; c < 4; ++c) {
            const bf16_t* kc = kl + (c * 16 + l15) * 64 + kbB;
            kf0[c] = load_frag16(kc,      kc + 16);
            kf1[c] = load_frag16(kc + 32, kc + 48);
        }
        v8f sc[4];
#pragma unroll
        for (int c = 0; c < 4; ++c) {
            sc[c] = {};
            sc[c] = wmma_bf16(aq0, kf0[c], sc[c]);
            sc[c] = wmma_bf16(aq1, kf1[c], sc[c]);
        }

        // --- online softmax (per row; rows live in 16-lane halves) ---
        float corr[8];
#pragma unroll
        for (int r = 0; r < 8; ++r) {
            float x0 = sc[0][r] * 0.125f, x1 = sc[1][r] * 0.125f;
            float x2 = sc[2][r] * 0.125f, x3 = sc[3][r] * 0.125f;
            float cm = fmaxf(fmaxf(x0, x1), fmaxf(x2, x3));
            cm = fmaxf(cm, __shfl_xor(cm, 1, 32));
            cm = fmaxf(cm, __shfl_xor(cm, 2, 32));
            cm = fmaxf(cm, __shfl_xor(cm, 4, 32));
            cm = fmaxf(cm, __shfl_xor(cm, 8, 32));
            float mnew = fmaxf(mrun[r], cm);
            float c  = __expf(mrun[r] - mnew);
            float p0 = __expf(x0 - mnew), p1 = __expf(x1 - mnew);
            float p2 = __expf(x2 - mnew), p3 = __expf(x3 - mnew);
            float ls = (p0 + p1) + (p2 + p3);
            ls += __shfl_xor(ls, 1, 32);
            ls += __shfl_xor(ls, 2, 32);
            ls += __shfl_xor(ls, 4, 32);
            ls += __shfl_xor(ls, 8, 32);
            lrun[r] = lrun[r] * c + ls;
            mrun[r] = mnew;
            corr[r] = c;
            bf16_t* prow = myp + (rbase + r) * 64 + l15;
            prow[0]  = (bf16_t)p0;
            prow[16] = (bf16_t)p1;
            prow[32] = (bf16_t)p2;
            prow[48] = (bf16_t)p3;
        }
#pragma unroll
        for (int t = 0; t < 4; ++t)
#pragma unroll
            for (int r = 0; r < 8; ++r) o[t][r] *= corr[r];

        // make P stores visible before A-fragment reload (same wave, LDS)
        asm volatile("s_wait_dscnt 0" ::: "memory");
        const bf16_t* prow = myp + l15 * 64;
        v16bf pf0 = load_frag16(prow + kbA,      prow + 16 + kbA);  // s 0..31
        v16bf pf1 = load_frag16(prow + 32 + kbA, prow + 48 + kbA);  // s 32..63

        // --- O += P (16x64) . V_chunk (64x64, LDS) --- (fragments hoisted)
        v16bf vf0[4], vf1[4];
#pragma unroll
        for (int t = 0; t < 4; ++t) {
            const bf16_t* vc = vl + (t * 16 + l15) * 64 + kbB;
            vf0[t] = load_frag16(vc,      vc + 16);
            vf1[t] = load_frag16(vc + 32, vc + 48);
        }
#pragma unroll
        for (int t = 0; t < 4; ++t) {
            o[t] = wmma_bf16(pf0, vf0[t], o[t]);
            o[t] = wmma_bf16(pf1, vf1[t], o[t]);
        }
        __syncthreads();
    }

    // --- normalize and store merged-head bf16 output ---
#pragma unroll
    for (int t = 0; t < 4; ++t)
#pragma unroll
        for (int r = 0; r < 8; ++r) {
            float val = o[t][r] / lrun[r];
            int trow = t0 + rbase + r;
            int n = qh * 64 + t * 16 + l15;
            AO[(size_t)(b * T_SEQ + trow) * 2048 + n] = (bf16_t)val;
        }
}

// ---------------------------------------------------------------------------
extern "C" void kernel_launch(void* const* d_in, const int* in_sizes, int n_in,
                              void* d_out, int out_size, void* d_ws, size_t ws_size,
                              hipStream_t stream) {
    (void)in_sizes; (void)n_in; (void)out_size; (void)ws_size;
    const float* x  = (const float*)d_in[0];
    const float* Wq = (const float*)d_in[1];
    const float* bq = (const float*)d_in[2];
    const float* Wk = (const float*)d_in[3];
    const float* bk = (const float*)d_in[4];
    const float* Wv = (const float*)d_in[5];
    const float* bv = (const float*)d_in[6];
    const float* Wo = (const float*)d_in[7];
    const float* bo = (const float*)d_in[8];
    float* out = (float*)d_out;

    char* ws = (char*)d_ws;
    size_t off = 0;
    auto walloc = [&](size_t bytes) {
        void* p = (void*)(ws + off);
        off += ((bytes + 255) / 256) * 256;
        return p;
    };
    bf16_t* xb    = (bf16_t*)walloc((size_t)2 * 2048 * 2048 * 2);  // [4096][2048]
    bf16_t* WqkvT = (bf16_t*)walloc((size_t)3072 * 2048 * 2);      // [3072][2048]
    bf16_t* WoT   = (bf16_t*)walloc((size_t)2048 * 2048 * 2);      // [2048][2048]
    float*  bqkv  = (float*) walloc((size_t)3072 * 4);
    bf16_t* Qb    = (bf16_t*)walloc((size_t)2 * 32 * 2048 * 64 * 2);
    bf16_t* Kb    = (bf16_t*)walloc((size_t)2 * 8 * 2048 * 64 * 2);
    bf16_t* Vtb   = (bf16_t*)walloc((size_t)2 * 8 * 64 * 2048 * 2);
    bf16_t* AOb   = (bf16_t*)walloc((size_t)2 * 2048 * 2048 * 2);

    int nx = 2 * 2048 * 2048;
    cvt_copy<<<(nx + 255) / 256, 256, 0, stream>>>(x, xb, nx);
    // Fused QKV weights -> [3072][2048] (rows: Wq cols | Wk cols | Wv cols)
    cvt_transpose_tiled<<<dim3(2048 / 32, 2048 / 32), dim3(32, 8), 0, stream>>>(
        Wq, WqkvT, 2048, 2048);
    cvt_transpose_tiled<<<dim3(512 / 32, 2048 / 32), dim3(32, 8), 0, stream>>>(
        Wk, WqkvT + (size_t)2048 * 2048, 2048, 512);
    cvt_transpose_tiled<<<dim3(512 / 32, 2048 / 32), dim3(32, 8), 0, stream>>>(
        Wv, WqkvT + (size_t)2560 * 2048, 2048, 512);
    cvt_transpose_tiled<<<dim3(2048 / 32, 2048 / 32), dim3(32, 8), 0, stream>>>(
        Wo, WoT, 2048, 2048);
    concat_bias<<<(3072 + 255) / 256, 256, 0, stream>>>(bq, bk, bv, bqkv);

    // Fused QKV projection: M=4096, N=3072, K=2048
    gemm_bf16<1><<<dim3(3072 / 128, 4096 / 256), 256, 0, stream>>>(
        xb, WqkvT, bqkv, 4096, 3072, 2048, nullptr, Qb, Kb, Vtb);

    // Fused attention: grid (T/128 row-tiles, B*Hq)
    attn_kernel<<<dim3(T_SEQ / 128, 2 * 32), 256, 0, stream>>>(Qb, Kb, Vtb, AOb);

    // Output projection -> fp32 d_out: M=4096, N=2048, K=2048
    gemm_bf16<0><<<dim3(2048 / 128, 4096 / 256), 256, 0, stream>>>(
        AOb, WoT, bo, 4096, 2048, 2048, out, nullptr, nullptr, nullptr);
}
